// VQVAE_57526791962808
// MI455X (gfx1250) — matrix operand order repaired and checked by reading, not compiled, not run on hardware
//
#include <hip/hip_runtime.h>

typedef _Float16 v16h __attribute__((ext_vector_type(16)));
typedef _Float16 v8h  __attribute__((ext_vector_type(8)));
typedef float    v8f  __attribute__((ext_vector_type(8)));

// Bulk async copy global->LDS using the CDNA5 async-tensor path (ASYNCcnt).
// Each issuing lane moves 16B from its global address to the LDS address in its VGPR.
// LDS byte address = low 32 bits of the generic pointer.
__device__ __forceinline__ void async_copy_to_lds(_Float16* ldsDst, const _Float16* gSrc,
                                                  int bytes, int tid) {
  for (int off = tid * 16; off < bytes; off += 256 * 16) {
    unsigned int l = (unsigned int)(unsigned long long)ldsDst + (unsigned int)off;
    const char* g = (const char*)gSrc + off;
    asm volatile("global_load_async_to_lds_b128 %0, %1, off" :: "v"(l), "v"(g) : "memory");
  }
  asm volatile("s_wait_asynccnt 0x0" ::: "memory");
  __syncthreads();
}

// ---------------- one-shot weight pre-swizzle into WMMA-B-layout f16 blobs
// wse3: [tap(9)][ntile(4)][lane(32)][e(16)]        (enc3, 32ci -> 64co)
// wsd1: [tap(9)][chunk(2)][ntile(2)][lane(32)][e(16)] (dec1, 64ci -> 32co)
// wsd2: [tap(9)][lane(32)][e(16)]                  (dec2, 32ci -> 16co)
__global__ void k_prep(const float* __restrict__ ew3, const float* __restrict__ dw1,
                       const float* __restrict__ dw2, _Float16* __restrict__ wse3,
                       _Float16* __restrict__ wsd1, _Float16* __restrict__ wsd2) {
  int p = blockIdx.x * 256 + threadIdx.x;
  if (p < 18432) {
    int i = p;
    int e = i & 15, ln = (i >> 4) & 31, j = (i >> 9) & 3, t = i >> 11;
    int ci = ((ln >= 16) ? 8 : 0) + (e < 8 ? e : e + 8);
    int co = j * 16 + (ln & 15);
    wse3[i] = (_Float16)ew3[(t * 32 + ci) * 64 + co];
  } else if (p < 36864) {
    int i = p - 18432;
    int e = i & 15, ln = (i >> 4) & 31, j = (i >> 9) & 1, ch = (i >> 10) & 1, t = i >> 11;
    int ci = ch * 32 + ((ln >= 16) ? 8 : 0) + (e < 8 ? e : e + 8);
    int co = j * 16 + (ln & 15);
    wsd1[i] = (_Float16)dw1[(t * 64 + ci) * 32 + co];
  } else if (p < 41472) {
    int i = p - 36864;
    int e = i & 15, ln = (i >> 4) & 31, t = i >> 9;
    int ci = ((ln >= 16) ? 8 : 0) + (e < 8 ? e : e + 8);
    int co = ln & 15;
    wsd2[i] = (_Float16)dw2[(t * 32 + ci) * 16 + co];
  }
}

// ---------------- enc conv1: 1->16, stride2, relu.  [64,128,128,1] -> [64,64,64,16] (f16)
__global__ void k_enc1(const float* __restrict__ x, const float* __restrict__ w,
                       const float* __restrict__ b, _Float16* __restrict__ h1) {
  int p = blockIdx.x * 256 + threadIdx.x;            // 262144 output pixels
  int n = p >> 12, oy = (p >> 6) & 63, ox = p & 63;
  float xv[9];
#pragma unroll
  for (int ky = 0; ky < 3; ++ky)
#pragma unroll
    for (int kx = 0; kx < 3; ++kx) {
      int iy = 2 * oy + ky, ix = 2 * ox + kx;        // SAME pad_lo = 0
      xv[ky * 3 + kx] = (iy < 128 && ix < 128) ? x[((size_t)(n * 128 + iy)) * 128 + ix] : 0.f;
    }
#pragma unroll
  for (int co = 0; co < 16; ++co) {
    float a = b[co];
#pragma unroll
    for (int t = 0; t < 9; ++t) a += xv[t] * w[t * 16 + co];
    h1[(size_t)p * 16 + co] = (_Float16)fmaxf(a, 0.f);
  }
}

// ---------------- enc conv2: 16->32, stride2, relu. [64,64,64,16] -> [64,32,32,32] (f16)
__global__ void k_enc2(const _Float16* __restrict__ h1, const float* __restrict__ w,
                       const float* __restrict__ b, _Float16* __restrict__ h2) {
  __shared__ float wl[4608];
  for (int i = threadIdx.x; i < 4608; i += 256) wl[i] = w[i];
  __syncthreads();
  int p = blockIdx.x * 256 + threadIdx.x;            // 65536
  int n = p >> 10, oy = (p >> 5) & 31, ox = p & 31;
  float acc[32];
#pragma unroll
  for (int co = 0; co < 32; ++co) acc[co] = b[co];
  for (int ky = 0; ky < 3; ++ky) { int iy = 2 * oy + ky; if (iy >= 64) continue;
    for (int kx = 0; kx < 3; ++kx) { int ix = 2 * ox + kx; if (ix >= 64) continue;
      const _Float16* xr = h1 + ((size_t)((n * 64 + iy) * 64 + ix)) * 16;
      const float* wr = wl + (ky * 3 + kx) * 16 * 32;
#pragma unroll
      for (int ci = 0; ci < 16; ++ci) { float v = (float)xr[ci];
#pragma unroll
        for (int co = 0; co < 32; ++co) acc[co] += v * wr[ci * 32 + co]; } } }
#pragma unroll
  for (int co = 0; co < 32; ++co) h2[(size_t)p * 32 + co] = (_Float16)fmaxf(acc[co], 0.f);
}

// ---------------- codebook -> f16 copy + per-code ||c||^2
__global__ void k_cb(const float* __restrict__ cb, _Float16* __restrict__ cbf,
                     float* __restrict__ cc) {
  int c = blockIdx.x * 256 + threadIdx.x;
  if (c >= 512) return;
  float s = 0.f;
#pragma unroll
  for (int k = 0; k < 64; ++k) { float v = cb[c * 64 + k]; cbf[c * 64 + k] = (_Float16)v; s += v * v; }
  cc[c] = s;
}

// ---------------- enc conv3 via WMMA: 32->64 stride1. M=65536, K=9*32, N=64.
__global__ void __launch_bounds__(256) k_enc3(const _Float16* __restrict__ h2,
        const _Float16* __restrict__ wse3, const float* __restrict__ b3,
        float* __restrict__ ze, _Float16* __restrict__ zef) {
  __shared__ __align__(32) _Float16 wl[9 * 4 * 32 * 16];       // 36KB, pre-swizzled B
  async_copy_to_lds(wl, wse3, 9 * 4 * 32 * 16 * 2, threadIdx.x);
  const int lane = threadIdx.x & 31, wave = threadIdx.x >> 5;
  const int tile = blockIdx.x * 8 + wave, p0 = tile * 16;
  const int n = p0 >> 10, rem = p0 & 1023, y = rem >> 5, x0 = rem & 31;
  const int hi = lane >> 4, m = lane & 15;
  v8f acc[4];
#pragma unroll
  for (int j = 0; j < 4; ++j) acc[j] = (v8f){0,0,0,0,0,0,0,0};
  const v16h* wall = (const v16h*)wl;
  for (int t = 0; t < 9; ++t) {
    int yy = y + t / 3 - 1;
    int ix = x0 + m + t % 3 - 1;
    v8h a0 = {0,0,0,0,0,0,0,0}, a1 = {0,0,0,0,0,0,0,0};
    if (yy >= 0 && yy < 32 && ix >= 0 && ix < 32) {
      const _Float16* src = h2 + ((size_t)((n * 32 + yy) * 32 + ix)) * 32;
      a0 = *(const v8h*)(src + hi * 8);            // A: K in [hi*8, hi*8+8)
      a1 = *(const v8h*)(src + 16 + hi * 8);       // A: K in [16+hi*8, ...)
    }
    v16h A;
#pragma unroll
    for (int e = 0; e < 8; ++e) { A[e] = a0[e]; A[8 + e] = a1[e]; }
#pragma unroll
    for (int j = 0; j < 4; ++j) {
      v16h B = wall[(t * 4 + j) * 32 + lane];
      acc[j] = __builtin_amdgcn_wmma_f32_16x16x32_f16(false, A, false, B,
                                                      (short)0, acc[j], false, false);
    }
  }
#pragma unroll
  for (int j = 0; j < 4; ++j) {
    int col = j * 16 + m;
    float bb = b3[col];
#pragma unroll
    for (int r = 0; r < 8; ++r) {                  // C: M = r + hi*8, N = lane&15
      int row = p0 + r + hi * 8;
      float v = acc[j][r] + bb;
      ze[(size_t)row * 64 + col] = v;
      zef[(size_t)row * 64 + col] = (_Float16)v;
    }
  }
}

// ---------------- VQ via WMMA: dot(z,c) for 16 pixels x 512 codes, argmin of ||c||^2 - 2*dot
__global__ void __launch_bounds__(256) k_vq(const _Float16* __restrict__ zef,
        const _Float16* __restrict__ cbf, const float* __restrict__ cc,
        const float* __restrict__ cb, float* __restrict__ zq, float* __restrict__ qz,
        _Float16* __restrict__ zqf) {
  __shared__ __align__(32) _Float16 cbl[512 * 64];             // 64KB codebook in LDS
  async_copy_to_lds(cbl, cbf, 512 * 64 * 2, threadIdx.x);
  const int lane = threadIdx.x & 31, wave = threadIdx.x >> 5;
  const int p0 = (blockIdx.x * 8 + wave) * 16;
  const int hi = lane >> 4, m = lane & 15;
  const _Float16* zr = zef + (size_t)(p0 + m) * 64;
  v8h a00 = *(const v8h*)(zr + hi * 16);
  v8h a01 = *(const v8h*)(zr + 8 + hi * 16);
  v8h a10 = *(const v8h*)(zr + 32 + hi * 16);
  v8h a11 = *(const v8h*)(zr + 40 + hi * 16);
  v16h A0, A1;
#pragma unroll
  for (int e = 0; e < 8; ++e) { A0[e] = a00[e]; A0[8+e] = a01[e]; A1[e] = a10[e]; A1[8+e] = a11[e]; }
  float best[8]; int bidx[8];
#pragma unroll
  for (int r = 0; r < 8; ++r) { best[r] = 3.4e38f; bidx[r] = 0; }
  for (int ct = 0; ct < 32; ++ct) {
    int code = ct * 16 + m;
    const _Float16* cr = cbl + (size_t)code * 64;
    v16h B0 = *(const v16h*)(cr + hi * 16);        // B rows are layout-contiguous (ds)
    v16h B1 = *(const v16h*)(cr + 32 + hi * 16);
    v8f acc = (v8f){0,0,0,0,0,0,0,0};
    acc = __builtin_amdgcn_wmma_f32_16x16x32_f16(false, A0, false, B0, (short)0, acc, false, false);
    acc = __builtin_amdgcn_wmma_f32_16x16x32_f16(false, A1, false, B1, (short)0, acc, false, false);
    float c2 = cc[code];
#pragma unroll
    for (int r = 0; r < 8; ++r) {
      float d = c2 - 2.f * acc[r];
      if (d < best[r] || (d == best[r] && code < bidx[r])) { best[r] = d; bidx[r] = code; }
    }
  }
  // butterfly argmin across the 16 code-lanes of each half (wave32)
#pragma unroll
  for (int mask = 8; mask >= 1; mask >>= 1) {
#pragma unroll
    for (int r = 0; r < 8; ++r) {
      float ov = __shfl_xor(best[r], mask, 32);
      int   oi = __shfl_xor(bidx[r], mask, 32);
      if (ov < best[r] || (ov == best[r] && oi < bidx[r])) { best[r] = ov; bidx[r] = oi; }
    }
  }
  if (m == 0) {
#pragma unroll
    for (int r = 0; r < 8; ++r) qz[p0 + hi * 8 + r] = (float)bidx[r];
  }
  // gather z_q rows cooperatively (lane writes 2 floats + 2 halves per row)
#pragma unroll
  for (int mm = 0; mm < 16; ++mm) {
    int id = (mm < 8) ? __shfl(bidx[mm & 7], 0, 32) : __shfl(bidx[mm & 7], 16, 32);
    float2 v = *(const float2*)(cb + (size_t)id * 64 + 2 * lane);
    *(float2*)(zq + (size_t)(p0 + mm) * 64 + 2 * lane) = v;
    zqf[(size_t)(p0 + mm) * 64 + 2 * lane]     = (_Float16)v.x;
    zqf[(size_t)(p0 + mm) * 64 + 2 * lane + 1] = (_Float16)v.y;
  }
}

// ---------------- dec convT1 via WMMA (sub-pixel classes): 64->32 stride2 + relu.
__global__ void __launch_bounds__(256) k_dec1(const _Float16* __restrict__ zqf,
        const _Float16* __restrict__ wsd1, const float* __restrict__ b,
        _Float16* __restrict__ g1) {
  __shared__ __align__(32) _Float16 wl[9 * 2 * 2 * 32 * 16];   // 36KB, pre-swizzled B
  async_copy_to_lds(wl, wsd1, 9 * 2 * 2 * 32 * 16 * 2, threadIdx.x);
  const int lane = threadIdx.x & 31, wave = threadIdx.x >> 5;
  const int tile = blockIdx.x * 8 + wave;        // 16384 tiles = 4 classes * 4096
  const int cls = tile >> 12, ey = cls >> 1, ex = cls & 1;
  const int t2 = tile & 4095;
  const int n = t2 >> 6, yp = (t2 >> 1) & 31, x0 = (t2 & 1) * 16;
  const int hi = lane >> 4, m = lane & 15;
  v8f acc[2];
#pragma unroll
  for (int j = 0; j < 2; ++j) acc[j] = (v8f){0,0,0,0,0,0,0,0};
  const v16h* wall = (const v16h*)wl;
#pragma unroll
  for (int ky = ey; ky < 3; ky += 2) {           // taps with (ky%2)==(oy%2)
    int iy = yp + ((ey - ky) >> 1);              // (oy-ky)/2
    if (iy < 0 || iy >= 32) continue;
#pragma unroll
    for (int kx = ex; kx < 3; kx += 2) {
      int ix = x0 + m + ((ex - kx) >> 1);
      bool ok = (ix >= 0 && ix < 32);
      const _Float16* src = zqf + ((size_t)((n * 32 + iy) * 32 + (ok ? ix : 0))) * 64;
      int t = ky * 3 + kx;
#pragma unroll
      for (int ch = 0; ch < 2; ++ch) {           // K chunks: ci [0,32), [32,64)
        v8h a0 = {0,0,0,0,0,0,0,0}, a1 = {0,0,0,0,0,0,0,0};
        if (ok) {
          a0 = *(const v8h*)(src + ch * 32 + hi * 8);
          a1 = *(const v8h*)(src + ch * 32 + 16 + hi * 8);
        }
        v16h A;
#pragma unroll
        for (int e = 0; e < 8; ++e) { A[e] = a0[e]; A[8 + e] = a1[e]; }
#pragma unroll
        for (int j = 0; j < 2; ++j) {
          v16h B = wall[((t * 2 + ch) * 2 + j) * 32 + lane];
          acc[j] = __builtin_amdgcn_wmma_f32_16x16x32_f16(false, A, false, B,
                                                          (short)0, acc[j], false, false);
        }
      }
    }
  }
  const int oy = 2 * yp + ey;
#pragma unroll
  for (int j = 0; j < 2; ++j) {
    int co = j * 16 + m;
    float bb = b[co];
#pragma unroll
    for (int r = 0; r < 8; ++r) {
      int xp = x0 + r + hi * 8, ox = 2 * xp + ex;
      g1[((size_t)((n * 64 + oy) * 64 + ox)) * 32 + co] = (_Float16)fmaxf(acc[j][r] + bb, 0.f);
    }
  }
}

// ---------------- dec convT2 via WMMA (sub-pixel classes): 32->16 stride2 + relu.
__global__ void __launch_bounds__(256) k_dec2(const _Float16* __restrict__ g1,
        const _Float16* __restrict__ wsd2, const float* __restrict__ b,
        _Float16* __restrict__ g2) {
  __shared__ __align__(32) _Float16 wl[9 * 32 * 16];           // 9KB, pre-swizzled B
  async_copy_to_lds(wl, wsd2, 9 * 32 * 16 * 2, threadIdx.x);
  const int lane = threadIdx.x & 31, wave = threadIdx.x >> 5;
  const int tile = blockIdx.x * 8 + wave;        // 65536 tiles = 4 classes * 16384
  const int cls = tile >> 14, ey = cls >> 1, ex = cls & 1;
  const int t2 = tile & 16383;
  const int n = t2 >> 8, yp = (t2 >> 2) & 63, x0 = (t2 & 3) * 16;
  const int hi = lane >> 4, m = lane & 15;
  v8f acc = (v8f){0,0,0,0,0,0,0,0};
  const v16h* wall = (const v16h*)wl;
#pragma unroll
  for (int ky = ey; ky < 3; ky += 2) {
    int iy = yp + ((ey - ky) >> 1);
    if (iy < 0 || iy >= 64) continue;
#pragma unroll
    for (int kx = ex; kx < 3; kx += 2) {
      int ix = x0 + m + ((ex - kx) >> 1);
      bool ok = (ix >= 0 && ix < 64);
      v8h a0 = {0,0,0,0,0,0,0,0}, a1 = {0,0,0,0,0,0,0,0};
      if (ok) {
        const _Float16* src = g1 + ((size_t)((n * 64 + iy) * 64 + ix)) * 32;
        a0 = *(const v8h*)(src + hi * 8);
        a1 = *(const v8h*)(src + 16 + hi * 8);
      }
      v16h A;
#pragma unroll
      for (int e = 0; e < 8; ++e) { A[e] = a0[e]; A[8 + e] = a1[e]; }
      v16h B = wall[(ky * 3 + kx) * 32 + lane];
      acc = __builtin_amdgcn_wmma_f32_16x16x32_f16(false, A, false, B,
                                                   (short)0, acc, false, false);
    }
  }
  const int oy = 2 * yp + ey;
  float bb = b[m];                               // co = lane&15
#pragma unroll
  for (int r = 0; r < 8; ++r) {
    int xp = x0 + r + hi * 8, ox = 2 * xp + ex;
    g2[((size_t)((n * 128 + oy) * 128 + ox)) * 16 + m] = (_Float16)fmaxf(acc[r] + bb, 0.f);
  }
}

// ---------------- dec convT3: 16->1 stride1. [64,128,128,16] -> logits [64,128,128,1]
__global__ void k_dec3(const _Float16* __restrict__ g2, const float* __restrict__ w,
                       const float* __restrict__ b, float* __restrict__ out) {
  int p = blockIdx.x * 256 + threadIdx.x;           // 1048576
  int n = p >> 14, oy = (p >> 7) & 127, ox = p & 127;
  float acc = b[0];
  for (int ky = 0; ky < 3; ++ky) { int iy = oy + 1 - ky; if (iy < 0 || iy >= 128) continue;
    for (int kx = 0; kx < 3; ++kx) { int ix = ox + 1 - kx; if (ix < 0 || ix >= 128) continue;
      const _Float16* xr = g2 + ((size_t)((n * 128 + iy) * 128 + ix)) * 16;
#pragma unroll
      for (int ci = 0; ci < 16; ++ci) acc += (float)xr[ci] * w[(ky * 3 + kx) * 16 + ci];
    } }
  out[p] = acc;
}

extern "C" void kernel_launch(void* const* d_in, const int* in_sizes, int n_in,
                              void* d_out, int out_size, void* d_ws, size_t ws_size,
                              hipStream_t stream) {
  const float* x   = (const float*)d_in[0];
  const float* ew1 = (const float*)d_in[1];
  const float* eb1 = (const float*)d_in[2];
  const float* ew2 = (const float*)d_in[3];
  const float* eb2 = (const float*)d_in[4];
  const float* ew3 = (const float*)d_in[5];
  const float* eb3 = (const float*)d_in[6];
  const float* cb  = (const float*)d_in[7];
  const float* dw1 = (const float*)d_in[8];
  const float* db1 = (const float*)d_in[9];
  const float* dw2 = (const float*)d_in[10];
  const float* db2 = (const float*)d_in[11];
  const float* dw3 = (const float*)d_in[12];
  const float* db3 = (const float*)d_in[13];

  float* out    = (float*)d_out;                 // (logits, z_e, z_q, q_z) flat
  float* logits = out;
  float* ze     = out + 1048576;
  float* zq     = out + 1048576 + 4194304;
  float* qz     = out + 1048576 + 4194304 + 4194304;

  // workspace: time-aliased regions. A[0,32MB): h1f16 then g2f16. B[32,48MB): h2f16 then g1f16.
  // C[48,56MB): z_e f16. D[56MB,+): codebook f16 + ||c||^2. E[57MB,8MB): z_q f16.
  // F[65MB,+): pre-swizzled WMMA-B weight blobs.
  char* base = (char*)d_ws;
  _Float16* h1f  = (_Float16*)(base);
  _Float16* g2f  = (_Float16*)(base);
  _Float16* h2f  = (_Float16*)(base + (size_t)32 * 1024 * 1024);
  _Float16* g1f  = (_Float16*)(base + (size_t)32 * 1024 * 1024);
  _Float16* zef  = (_Float16*)(base + (size_t)48 * 1024 * 1024);
  _Float16* cbf  = (_Float16*)(base + (size_t)56 * 1024 * 1024);
  float*    cc   = (float*)   (base + (size_t)56 * 1024 * 1024 + 65536 * sizeof(_Float16));
  _Float16* zqf  = (_Float16*)(base + (size_t)57 * 1024 * 1024);
  _Float16* wse3 = (_Float16*)(base + (size_t)65 * 1024 * 1024);
  _Float16* wsd1 = wse3 + 18432;
  _Float16* wsd2 = wsd1 + 18432;

  k_prep<<<162, 256, 0, stream>>>(ew3, dw1, dw2, wse3, wsd1, wsd2);
  k_enc1<<<1024, 256, 0, stream>>>(x, ew1, eb1, h1f);
  k_enc2<<<256,  256, 0, stream>>>(h1f, ew2, eb2, h2f);
  k_cb  <<<2,    256, 0, stream>>>(cb, cbf, cc);
  k_enc3<<<512,  256, 0, stream>>>(h2f, wse3, eb3, ze, zef);
  k_vq  <<<512,  256, 0, stream>>>(zef, cbf, cc, cb, zq, qz, zqf);
  k_dec1<<<2048, 256, 0, stream>>>(zqf, wsd1, db1, g1f);
  k_dec2<<<8192, 256, 0, stream>>>(g1f, wsd2, db2, g2f);
  k_dec3<<<4096, 256, 0, stream>>>(g2f, dw3, db3, logits);
}
